// AutoRegressive_86655260164357
// MI455X (gfx1250) — matrix operand order repaired and checked by reading, not compiled, not run on hardware
//
#include <hip/hip_runtime.h>

// ---------------------------------------------------------------------------
// Feature detection for CDNA5 async global->LDS copies
// ---------------------------------------------------------------------------
#if defined(__has_builtin)
#  if __has_builtin(__builtin_amdgcn_global_load_async_to_lds_b128)
#    define HAVE_ASYNC 1
#  else
#    define HAVE_ASYNC 0
#  endif
#  if __has_builtin(__builtin_amdgcn_s_wait_asynccnt)
#    define HAVE_WAIT_ASYNC 1
#  else
#    define HAVE_WAIT_ASYNC 0
#  endif
#else
#  define HAVE_ASYNC 0
#  define HAVE_WAIT_ASYNC 0
#endif

// ---------------------------------------------------------------------------
// Types for CDNA5 WMMA
// ---------------------------------------------------------------------------
typedef __attribute__((ext_vector_type(16))) __bf16 v16bf;
typedef __attribute__((ext_vector_type(8)))  float  v8f;
typedef int v4i __attribute__((__vector_size__(4 * sizeof(int))));

__device__ __forceinline__ unsigned short f2bf(float f) {
  unsigned int u = __float_as_uint(f);
  u += 0x7FFFu + ((u >> 16) & 1u);   // round to nearest even
  return (unsigned short)(u >> 16);
}

#if HAVE_ASYNC
typedef __attribute__((address_space(1))) v4i GV4;  // global int4*
typedef __attribute__((address_space(3))) v4i LV4;  // LDS int4*
__device__ __forceinline__ void async_cp16(unsigned short* lds, const unsigned short* g) {
  v4i* gp = (v4i*)g;          // generic pointee retype
  v4i* lp = (v4i*)lds;
  __builtin_amdgcn_global_load_async_to_lds_b128((GV4*)gp, (LV4*)lp, 0, 0);
}
#endif

template <int N>
__device__ __forceinline__ void wait_async() {
#if HAVE_WAIT_ASYNC
  __builtin_amdgcn_s_wait_asynccnt(N);
#else
  if (N == 0) asm volatile("s_wait_asynccnt 0x0" ::: "memory");
  else        asm volatile("s_wait_asynccnt 0x3" ::: "memory");
#endif
}

// ---------------------------------------------------------------------------
// fp32 -> bf16 conversion
// ---------------------------------------------------------------------------
__global__ __launch_bounds__(256) void f32_to_bf16_kernel(const float* __restrict__ in,
                                                          unsigned short* __restrict__ out,
                                                          long long n) {
  long long i = (long long)blockIdx.x * 256 + threadIdx.x;
  long long stride = (long long)gridDim.x * 256;
  for (; i < n; i += stride) out[i] = f2bf(in[i]);
}

// ---------------------------------------------------------------------------
// Embedding + positional encoding
// ---------------------------------------------------------------------------
__global__ __launch_bounds__(256) void embed_kernel(const int* __restrict__ text,
                                                    const int* __restrict__ audio,
                                                    const int* __restrict__ tlb,
                                                    const int* __restrict__ alb,
                                                    const float* __restrict__ ttab,
                                                    const float* __restrict__ atab,
                                                    float* __restrict__ x,
                                                    unsigned short* __restrict__ memb,
                                                    int L, int Tt, int Ta, int D) {
  const int bl  = blockIdx.x;
  const int b   = bl / L;
  const int pos = bl % L;
  const int tl  = tlb[b];
  const int al  = alb[b];
  float* xr = x + (long long)bl * D;
  unsigned short* mr = memb + (long long)bl * D;

  const float* tab = nullptr;
  int p = 0;
  if (pos < tl) { p = pos; tab = ttab + (long long)text[b * Tt + p] * D; }
  else if (pos < tl + al) { p = pos - tl; tab = atab + (long long)audio[b * Ta + p] * D; }

  #pragma unroll
  for (int i = 0; i < 4; ++i) {
    const int d = threadIdx.x + i * 256;
    float val = 0.f;
    if (tab) {
      const float div = __expf(-(float)(d & ~1) * (9.210340371976184f / (float)D));
      const float ang = (float)p * div;
      const float pe  = (d & 1) ? __cosf(ang) : __sinf(ang);
      val = tab[d] + pe;
    }
    xr[d] = val;
    mr[d] = f2bf(val);
  }
}

// ---------------------------------------------------------------------------
// LayerNorm -> bf16
// ---------------------------------------------------------------------------
__global__ __launch_bounds__(256) void layernorm_bf16_kernel(const float* __restrict__ X,
                                                             const float* __restrict__ w,
                                                             const float* __restrict__ b,
                                                             unsigned short* __restrict__ out,
                                                             int D) {
  __shared__ float s1[256];
  __shared__ float s2[256];
  const long long row = blockIdx.x;
  const float* x = X + row * D;
  float v[4];
  float a0 = 0.f, a1 = 0.f;
  #pragma unroll
  for (int i = 0; i < 4; ++i) {
    v[i] = x[threadIdx.x + i * 256];
    a0 += v[i];
    a1 += v[i] * v[i];
  }
  s1[threadIdx.x] = a0; s2[threadIdx.x] = a1;
  __syncthreads();
  for (int st = 128; st > 0; st >>= 1) {
    if (threadIdx.x < st) { s1[threadIdx.x] += s1[threadIdx.x + st]; s2[threadIdx.x] += s2[threadIdx.x + st]; }
    __syncthreads();
  }
  const float mu   = s1[0] / (float)D;
  const float var  = s2[0] / (float)D - mu * mu;
  const float rstd = rsqrtf(var + 1e-5f);
  #pragma unroll
  for (int i = 0; i < 4; ++i) {
    const int d = threadIdx.x + i * 256;
    out[row * D + d] = f2bf((v[i] - mu) * rstd * w[d] + b[d]);
  }
}

// ---------------------------------------------------------------------------
// Row softmax (wave32 per row), fp32 in -> bf16 probabilities out
// ---------------------------------------------------------------------------
__global__ __launch_bounds__(256) void softmax_rows_kernel(const float* __restrict__ S,
                                                           unsigned short* __restrict__ P,
                                                           long long rows, int cols) {
  const long long row = (long long)blockIdx.x * 8 + (threadIdx.x >> 5);
  const int lane = threadIdx.x & 31;
  if (row >= rows) return;
  const float* s = S + row * cols;
  float m = -__builtin_inff();
  for (int c = lane; c < cols; c += 32) m = fmaxf(m, s[c]);
  #pragma unroll
  for (int off = 16; off; off >>= 1) m = fmaxf(m, __shfl_xor(m, off, 32));
  float sum = 0.f;
  for (int c = lane; c < cols; c += 32) sum += __expf(s[c] - m);
  #pragma unroll
  for (int off = 16; off; off >>= 1) sum += __shfl_xor(sum, off, 32);
  const float inv = 1.0f / sum;
  unsigned short* pr = P + row * cols;
  for (int c = lane; c < cols; c += 32) pr[c] = f2bf(__expf(s[c] - m) * inv);
}

// ---------------------------------------------------------------------------
// WMMA GEMM: C = A(bf16, MxK row-major) * B(bf16, stored [N][K] k-contiguous)^T
// Block tile 128x64x32, 8 waves, each wave 32x32 of C via 4x v_wmma_f32_16x16x32_bf16.
// Async global->LDS double-buffered pipeline on CDNA5.
// ---------------------------------------------------------------------------
enum { EPI_F32 = 0, EPI_BF16_RELU = 1, EPI_SCORES = 2, EPI_QKV = 3, EPI_CTX = 4 };

struct GemmParams {
  const unsigned short* A;  long long lda;  long long batchA;
  const unsigned short* Bm; long long ldb;  long long batchB;
  int M, N, K;
  float* Cf;
  unsigned short* Cb;
  long long ldc; long long batchC;
  const float* bias;
  const float* residual; long long ldr;
  float scale;
  const int* tl; const int* al;
  int H, L, Dm, noff;
  unsigned short* qb; unsigned short* kb; unsigned short* vtb;
};

#define ASTRIDE (128 * 32)
#define BSTRIDE (64 * 32)

template <int EPI>
__global__ __launch_bounds__(256) void wmma_gemm(GemmParams p) {
  __shared__ __align__(16) unsigned short As[2 * ASTRIDE];  // [buf][m][k]
  __shared__ __align__(16) unsigned short Bs[2 * BSTRIDE];  // [buf][n][k]
  const int bz    = blockIdx.z;
  const int tileM = blockIdx.y * 128;
  const int tileN = blockIdx.x * 64;
  const unsigned short* A  = p.A  + (long long)bz * p.batchA;
  const unsigned short* Bw = p.Bm + (long long)bz * p.batchB;
  const int t    = threadIdx.x;
  const int lane = t & 31;
  const int wave = t >> 5;
  const int wm   = (wave & 3) * 32;  // wave M offset (0,32,64,96)
  const int wn   = (wave >> 2) * 32; // wave N offset (0,32)
  const int lh   = lane >> 4;        // lane half
  const int l16  = lane & 15;
  const int ldRow = t >> 2;          // 0..63
  const int ldK   = (t & 3) * 8;     // 0,8,16,24
  int ldNB = tileN + ldRow;          // clamped B row (always issue)
  if (ldNB > p.N - 1) ldNB = p.N - 1;

  v8f acc00 = {}, acc01 = {}, acc10 = {}, acc11 = {};

  auto compute_stage = [&](int buf) {
    const unsigned short* as = As + buf * ASTRIDE;
    const unsigned short* bs = Bs + buf * BSTRIDE;
    union { v16bf v; uint4 q[2]; } fa0, fa1, fb0, fb1;
    // A frag: lanes 0-15 rows M, K{0..7,16..23}; lanes 16-31 K{8..15,24..31}
    const unsigned short* ap = &as[(wm + l16) * 32 + lh * 8];
    fa0.q[0] = *(const uint4*)(ap);
    fa0.q[1] = *(const uint4*)(ap + 16);
    fa1.q[0] = *(const uint4*)(ap + 512);        // +16 rows
    fa1.q[1] = *(const uint4*)(ap + 512 + 16);
    // B frag: lanes 0-15 -> N, K{0..7,16..23}; lanes 16-31 -> K{8..15,24..31}
    const unsigned short* bp = &bs[(wn + l16) * 32 + lh * 8];
    fb0.q[0] = *(const uint4*)(bp);
    fb0.q[1] = *(const uint4*)(bp + 16);
    fb1.q[0] = *(const uint4*)(bp + 512);        // +16 cols
    fb1.q[1] = *(const uint4*)(bp + 512 + 16);
    acc00 = __builtin_amdgcn_wmma_f32_16x16x32_bf16(false, fa0.v, false, fb0.v, (short)0, acc00, false, false);
    acc01 = __builtin_amdgcn_wmma_f32_16x16x32_bf16(false, fa0.v, false, fb1.v, (short)0, acc01, false, false);
    acc10 = __builtin_amdgcn_wmma_f32_16x16x32_bf16(false, fa1.v, false, fb0.v, (short)0, acc10, false, false);
    acc11 = __builtin_amdgcn_wmma_f32_16x16x32_bf16(false, fa1.v, false, fb1.v, (short)0, acc11, false, false);
  };

#if HAVE_ASYNC
  auto issue_stage = [&](int k0, int buf) {
    unsigned short* as = As + buf * ASTRIDE;
    unsigned short* bs = Bs + buf * BSTRIDE;
    // A tile: 128x32, each thread two 16B async copies
    async_cp16(&as[ldRow * 32 + ldK],
               (const unsigned short*)&A[(long long)(tileM + ldRow) * p.lda + k0 + ldK]);
    async_cp16(&as[(ldRow + 64) * 32 + ldK],
               (const unsigned short*)&A[(long long)(tileM + ldRow + 64) * p.lda + k0 + ldK]);
    // B tile: 64x32, one 16B async copy (row clamped so every wave issues 3 ops)
    async_cp16(&bs[ldRow * 32 + ldK],
               (const unsigned short*)&Bw[(long long)ldNB * p.ldb + k0 + ldK]);
  };
  issue_stage(0, 0);
  int buf = 0;
  for (int k0 = 0; k0 < p.K; k0 += 32) {
    const bool more = (k0 + 32 < p.K);
    if (more) {
      issue_stage(k0 + 32, buf ^ 1);
      wait_async<3>();   // the 3 just-issued may remain; previous stage retired (in-order)
    } else {
      wait_async<0>();
    }
    __syncthreads();
    compute_stage(buf);
    __syncthreads();
    buf ^= 1;
  }
#else
  for (int k0 = 0; k0 < p.K; k0 += 32) {
    const uint4 va0 = *(const uint4*)(&A[(long long)(tileM + ldRow) * p.lda + k0 + ldK]);
    const uint4 va1 = *(const uint4*)(&A[(long long)(tileM + ldRow + 64) * p.lda + k0 + ldK]);
    const uint4 vb  = *(const uint4*)(&Bw[(long long)ldNB * p.ldb + k0 + ldK]);
    if (k0 + 32 < p.K) {
      __builtin_prefetch(&A[(long long)(tileM + ldRow) * p.lda + k0 + 32 + ldK], 0, 1);
      __builtin_prefetch(&Bw[(long long)ldNB * p.ldb + k0 + 32 + ldK], 0, 1);
    }
    __syncthreads();
    *(uint4*)(&As[ldRow * 32 + ldK]) = va0;
    *(uint4*)(&As[(ldRow + 64) * 32 + ldK]) = va1;
    *(uint4*)(&Bs[ldRow * 32 + ldK]) = vb;
    __syncthreads();
    compute_stage(0);
  }
#endif

  // Epilogue.  C layout: col = l16 (+16 for *1 accs), row = wm + mi*16 + lh*8 + r
  #pragma unroll
  for (int mi = 0; mi < 2; ++mi) {
    #pragma unroll
    for (int r = 0; r < 8; ++r) {
      const int m = tileM + wm + mi * 16 + lh * 8 + r;
      #pragma unroll
      for (int half = 0; half < 2; ++half) {
        const int n = tileN + wn + half * 16 + l16;
        if (n >= p.N) continue;
        float v;
        if (mi == 0) v = half ? acc01[r] : acc00[r];
        else         v = half ? acc11[r] : acc10[r];
        if (EPI == EPI_F32) {
          if (p.bias)     v += p.bias[n];
          if (p.residual) v += p.residual[(long long)m * p.ldr + n];
          p.Cf[(long long)bz * p.batchC + (long long)m * p.ldc + n] = v;
        } else if (EPI == EPI_BF16_RELU) {
          v += p.bias[n];
          v = v > 0.f ? v : 0.f;
          p.Cb[(long long)m * p.ldc + n] = f2bf(v);
        } else if (EPI == EPI_SCORES) {
          bool allowed = true;
          if (p.tl) {
            const int b  = bz / p.H;
            const int tv = p.tl[b], av = p.al[b];
            allowed = (n < tv) || ((n <= m) && (m < tv + av));
          }
          p.Cf[(long long)bz * p.batchC + (long long)m * p.ldc + n] =
              allowed ? v * p.scale : -__builtin_inff();
        } else if (EPI == EPI_QKV) {
          const int b   = m / p.L;
          const int pos = m % p.L;
          const int ng  = n + p.noff;
          v += p.bias[ng];
          const int sel = ng >> 10;
          const int hh  = (ng & 1023) >> 6;
          const int d   = ng & 63;
          const unsigned short bf = f2bf(v);
          const long long qk = (((long long)(b * p.H + hh)) * p.L + pos) * 64 + d;
          if (sel == 0)      p.qb[qk] = bf;
          else if (sel == 1) p.kb[qk] = bf;
          else p.vtb[(((long long)(b * p.H + hh)) * 64 + d) * p.L + pos] = bf;  // V^T
        } else if (EPI == EPI_CTX) {
          const int b  = bz / p.H;
          const int hh = bz % p.H;
          p.Cb[((long long)(b * p.L + m)) * p.ldc + hh * 64 + n] = f2bf(v);
        }
      }
    }
  }
}

// ---------------------------------------------------------------------------
// Host side
// ---------------------------------------------------------------------------
static void gemm_dispatch(int epi, dim3 grid, const GemmParams& p, hipStream_t s) {
  switch (epi) {
    case EPI_F32:       wmma_gemm<EPI_F32><<<grid, 256, 0, s>>>(p); break;
    case EPI_BF16_RELU: wmma_gemm<EPI_BF16_RELU><<<grid, 256, 0, s>>>(p); break;
    case EPI_SCORES:    wmma_gemm<EPI_SCORES><<<grid, 256, 0, s>>>(p); break;
    case EPI_QKV:       wmma_gemm<EPI_QKV><<<grid, 256, 0, s>>>(p); break;
    case EPI_CTX:       wmma_gemm<EPI_CTX><<<grid, 256, 0, s>>>(p); break;
  }
}

extern "C" void kernel_launch(void* const* d_in, const int* in_sizes, int n_in,
                              void* d_out, int out_size, void* d_ws, size_t ws_size,
                              hipStream_t stream) {
  (void)in_sizes; (void)n_in; (void)out_size; (void)ws_size;
  const int B = 4, Tt = 128, Ta = 640, D = 1024, DFF = 4096, NL = 6, H = 16, VA = 1026;
  const int L = Tt + Ta;      // 768
  const int M = B * L;        // 3072
  const int dh = 64;

  const int*   text   = (const int*)d_in[0];
  const int*   audio  = (const int*)d_in[1];
  const int*   tl     = (const int*)d_in[2];
  const int*   al     = (const int*)d_in[3];
  const float* ttab   = (const float*)d_in[4];
  const float* atab   = (const float*)d_in[5];
  const float* sa_in_w  = (const float*)d_in[6];
  const float* sa_in_b  = (const float*)d_in[7];
  const float* sa_out_w = (const float*)d_in[8];
  const float* sa_out_b = (const float*)d_in[9];
  const float* ca_in_w  = (const float*)d_in[10];
  const float* ca_in_b  = (const float*)d_in[11];
  const float* ca_out_w = (const float*)d_in[12];
  const float* ca_out_b = (const float*)d_in[13];
  const float* ln1_w = (const float*)d_in[14];
  const float* ln1_b = (const float*)d_in[15];
  const float* ln2_w = (const float*)d_in[16];
  const float* ln2_b = (const float*)d_in[17];
  const float* ln3_w = (const float*)d_in[18];
  const float* ln3_b = (const float*)d_in[19];
  const float* ffn_w1 = (const float*)d_in[20];
  const float* ffn_b1 = (const float*)d_in[21];
  const float* ffn_w2 = (const float*)d_in[22];
  const float* ffn_b2 = (const float*)d_in[23];
  float* out = (float*)d_out;

  char* ws = (char*)d_ws;
  auto alloc = [&](long long bytes) -> char* {
    char* r = ws;
    ws += (bytes + 255) & ~255LL;
    return r;
  };
  unsigned short* w_sa_in  = (unsigned short*)alloc((long long)NL * 3 * D * D * 2);
  unsigned short* w_sa_out = (unsigned short*)alloc((long long)NL * D * D * 2);
  unsigned short* w_ca_in  = (unsigned short*)alloc((long long)NL * 3 * D * D * 2);
  unsigned short* w_ca_out = (unsigned short*)alloc((long long)NL * D * D * 2);
  unsigned short* w_ffn1   = (unsigned short*)alloc((long long)NL * DFF * D * 2);
  unsigned short* w_ffn2   = (unsigned short*)alloc((long long)NL * D * DFF * 2);
  unsigned short* w_atab   = (unsigned short*)alloc((long long)VA * D * 2);
  float*          x        = (float*)alloc((long long)M * D * 4);
  unsigned short* memb     = (unsigned short*)alloc((long long)M * D * 2);
  unsigned short* hb       = (unsigned short*)alloc((long long)M * D * 2);
  unsigned short* qb       = (unsigned short*)alloc((long long)B * H * L * dh * 2);
  unsigned short* kb       = (unsigned short*)alloc((long long)B * H * L * dh * 2);
  unsigned short* vtb      = (unsigned short*)alloc((long long)B * H * dh * L * 2);
  unsigned short* ctxb     = (unsigned short*)alloc((long long)M * D * 2);
  unsigned short* ffb      = (unsigned short*)alloc((long long)M * DFF * 2);
  float*          S        = (float*)alloc((long long)B * H * L * L * 4);
  unsigned short* P        = (unsigned short*)alloc((long long)B * H * L * L * 2);

  auto conv = [&](const float* src, unsigned short* dst, long long n) {
    int blocks = (int)((n + 255) / 256);
    if (blocks > 4096) blocks = 4096;
    f32_to_bf16_kernel<<<blocks, 256, 0, stream>>>(src, dst, n);
  };
  conv(sa_in_w,  w_sa_in,  (long long)NL * 3 * D * D);
  conv(sa_out_w, w_sa_out, (long long)NL * D * D);
  conv(ca_in_w,  w_ca_in,  (long long)NL * 3 * D * D);
  conv(ca_out_w, w_ca_out, (long long)NL * D * D);
  conv(ffn_w1,   w_ffn1,   (long long)NL * DFF * D);
  conv(ffn_w2,   w_ffn2,   (long long)NL * D * DFF);
  conv(atab,     w_atab,   (long long)VA * D);

  embed_kernel<<<M, 256, 0, stream>>>(text, audio, tl, al, ttab, atab, x, memb, L, Tt, Ta, D);

  const float scale = 0.125f;  // 1/sqrt(64)

  for (int l = 0; l < NL; ++l) {
    // ---------------- self attention ----------------
    layernorm_bf16_kernel<<<M, 256, 0, stream>>>(x, ln1_w + l * D, ln1_b + l * D, hb, D);
    {
      GemmParams p{};
      p.A = hb; p.lda = D;
      p.Bm = w_sa_in + (long long)l * 3 * D * D; p.ldb = D;
      p.M = M; p.N = 3 * D; p.K = D;
      p.bias = sa_in_b + (long long)l * 3 * D;
      p.qb = qb; p.kb = kb; p.vtb = vtb; p.H = H; p.L = L; p.noff = 0;
      gemm_dispatch(EPI_QKV, dim3(48, 24, 1), p, stream);
    }
    {
      GemmParams p{};
      p.A = qb; p.lda = dh; p.batchA = (long long)L * dh;
      p.Bm = kb; p.ldb = dh; p.batchB = (long long)L * dh;
      p.M = L; p.N = L; p.K = dh;
      p.Cf = S; p.ldc = L; p.batchC = (long long)L * L;
      p.scale = scale; p.tl = tl; p.al = al; p.H = H;
      gemm_dispatch(EPI_SCORES, dim3(12, 6, B * H), p, stream);
    }
    softmax_rows_kernel<<<(B * H * L + 7) / 8, 256, 0, stream>>>(S, P, (long long)B * H * L, L);
    {
      GemmParams p{};
      p.A = P; p.lda = L; p.batchA = (long long)L * L;
      p.Bm = vtb; p.ldb = L; p.batchB = (long long)dh * L;
      p.M = L; p.N = dh; p.K = L;
      p.Cb = ctxb; p.ldc = D; p.H = H; p.L = L;
      gemm_dispatch(EPI_CTX, dim3(1, 6, B * H), p, stream);
    }
    {
      GemmParams p{};
      p.A = ctxb; p.lda = D;
      p.Bm = w_sa_out + (long long)l * D * D; p.ldb = D;
      p.M = M; p.N = D; p.K = D;
      p.Cf = x; p.ldc = D;
      p.bias = sa_out_b + (long long)l * D;
      p.residual = x; p.ldr = D;
      gemm_dispatch(EPI_F32, dim3(16, 24, 1), p, stream);
    }

    // ---------------- cross attention ----------------
    layernorm_bf16_kernel<<<M, 256, 0, stream>>>(x, ln2_w + l * D, ln2_b + l * D, hb, D);
    {
      GemmParams p{};  // Q projection from LN(x)
      p.A = hb; p.lda = D;
      p.Bm = w_ca_in + (long long)l * 3 * D * D; p.ldb = D;
      p.M = M; p.N = D; p.K = D;
      p.bias = ca_in_b + (long long)l * 3 * D;
      p.qb = qb; p.kb = kb; p.vtb = vtb; p.H = H; p.L = L; p.noff = 0;
      gemm_dispatch(EPI_QKV, dim3(16, 24, 1), p, stream);
    }
    {
      GemmParams p{};  // K/V projection from memory (initial embeddings)
      p.A = memb; p.lda = D;
      p.Bm = w_ca_in + (long long)l * 3 * D * D + (long long)D * D; p.ldb = D;
      p.M = M; p.N = 2 * D; p.K = D;
      p.bias = ca_in_b + (long long)l * 3 * D;
      p.qb = qb; p.kb = kb; p.vtb = vtb; p.H = H; p.L = L; p.noff = D;
      gemm_dispatch(EPI_QKV, dim3(32, 24, 1), p, stream);
    }
    {
      GemmParams p{};  // scores, no mask
      p.A = qb; p.lda = dh; p.batchA = (long long)L * dh;
      p.Bm = kb; p.ldb = dh; p.batchB = (long long)L * dh;
      p.M = L; p.N = L; p.K = dh;
      p.Cf = S; p.ldc = L; p.batchC = (long long)L * L;
      p.scale = scale; p.tl = nullptr; p.al = nullptr; p.H = H;
      gemm_dispatch(EPI_SCORES, dim3(12, 6, B * H), p, stream);
    }
    softmax_rows_kernel<<<(B * H * L + 7) / 8, 256, 0, stream>>>(S, P, (long long)B * H * L, L);
    {
      GemmParams p{};
      p.A = P; p.lda = L; p.batchA = (long long)L * L;
      p.Bm = vtb; p.ldb = L; p.batchB = (long long)dh * L;
      p.M = L; p.N = dh; p.K = L;
      p.Cb = ctxb; p.ldc = D; p.H = H; p.L = L;
      gemm_dispatch(EPI_CTX, dim3(1, 6, B * H), p, stream);
    }
    {
      GemmParams p{};
      p.A = ctxb; p.lda = D;
      p.Bm = w_ca_out + (long long)l * D * D; p.ldb = D;
      p.M = M; p.N = D; p.K = D;
      p.Cf = x; p.ldc = D;
      p.bias = ca_out_b + (long long)l * D;
      p.residual = x; p.ldr = D;
      gemm_dispatch(EPI_F32, dim3(16, 24, 1), p, stream);
    }

    // ---------------- FFN ----------------
    layernorm_bf16_kernel<<<M, 256, 0, stream>>>(x, ln3_w + l * D, ln3_b + l * D, hb, D);
    {
      GemmParams p{};
      p.A = hb; p.lda = D;
      p.Bm = w_ffn1 + (long long)l * DFF * D; p.ldb = D;
      p.M = M; p.N = DFF; p.K = D;
      p.Cb = ffb; p.ldc = DFF;
      p.bias = ffn_b1 + (long long)l * DFF;
      gemm_dispatch(EPI_BF16_RELU, dim3(64, 24, 1), p, stream);
    }
    {
      GemmParams p{};
      p.A = ffb; p.lda = DFF;
      p.Bm = w_ffn2 + (long long)l * D * DFF; p.ldb = DFF;
      p.M = M; p.N = D; p.K = DFF;
      p.Cf = x; p.ldc = D;
      p.bias = ffn_b2 + (long long)l * D;
      p.residual = x; p.ldr = D;
      gemm_dispatch(EPI_F32, dim3(16, 24, 1), p, stream);
    }
  }

  // ---------------- final projection: logits = x @ audio_table^T ----------------
  conv(x, hb, (long long)M * D);
  {
    GemmParams p{};
    p.A = hb; p.lda = D;
    p.Bm = w_atab; p.ldb = D;
    p.M = M; p.N = VA; p.K = D;
    p.Cf = out; p.ldc = VA;
    gemm_dispatch(EPI_F32, dim3((VA + 63) / 64, 24, 1), p, stream);
  }
}
